// SparseSamplingAttention_87462714016292
// MI455X (gfx1250) — matrix operand order repaired and checked by reading, not compiled, not run on hardware
//
#include <hip/hip_runtime.h>
#include <hip/hip_bf16.h>

typedef _Float16 f16;
typedef __bf16   bf16;
typedef __attribute__((ext_vector_type(16))) _Float16 v16h;
typedef __attribute__((ext_vector_type(8)))  _Float16 v8h;
typedef __attribute__((ext_vector_type(16))) __bf16   v16bf;
typedef __attribute__((ext_vector_type(8)))  float    v8f;

#define B_    8
#define C_    256
#define H_    128
#define W_    128
#define HW_   16384
#define NH_   8
#define HD_   32
#define WS_   8
#define WN_   16
#define NWIN_ 256
#define WSQ_  64
#define N_    64          // B_*NH_

// ---------------- WMMA helpers (CDNA5 gfx1250, wave32) ----------------

__device__ inline v8f zero8() {
  v8f z;
#pragma unroll
  for (int i = 0; i < 8; ++i) z[i] = 0.f;
  return z;
}

__device__ inline v8f wmma_f16(v16h a, v16h b, v8f c) {
  return __builtin_amdgcn_wmma_f32_16x16x32_f16(false, a, false, b, (short)0, c, false, false);
}
__device__ inline v8f wmma_bf16(v16bf a, v16bf b, v8f c) {
  return __builtin_amdgcn_wmma_f32_16x16x32_bf16(false, a, false, b, (short)0, c, false, false);
}

// A-fragment (16x32 f16, row-major source, ld in halves).
// Contiguous order used here (matches loadBT16): lanes 0-15 hold K {0..15}, lanes 16-31 hold K {16..31}.
__device__ inline v16h loadA16(const f16* base, int ld, int lane) {
  int row = lane & 15;
  int off = (lane >> 4) * 16;
  const f16* r = base + row * ld;
  v8h lo = *(const v8h*)(r + off);
  v8h hi = *(const v8h*)(r + off + 8);
  v16h a;
#pragma unroll
  for (int i = 0; i < 8; ++i) { a[i] = lo[i]; a[8 + i] = hi[i]; }
  return a;
}

// B-fragment (32x16 f16) from a K-contiguous, N-major buffer bT[n][K] (ldK halves).
// Per lane: n = lane&15, contiguous K run starting at 16*(lane>>4).
__device__ inline v16h loadBT16(const f16* bT, int ldK, int lane) {
  const f16* p = bT + (lane & 15) * ldK + (lane >> 4) * 16;
  return *(const v16h*)p;
}

// ---------------- weight swizzle: fp32 [M,Cin] -> bf16 A-fragment stream ----------------
// frag f = mt*Kc + kc, 512 halves per fragment, element (lane,h) at f*512+lane*16+h.
__global__ void swizzle_w_kernel(const float* __restrict__ w, bf16* __restrict__ out,
                                 int Mtiles, int Kc, int Cin) {
  int idx = blockIdx.x * blockDim.x + threadIdx.x;
  int total = Mtiles * Kc * 512;
  if (idx >= total) return;
  int h    = idx & 15;
  int lane = (idx >> 4) & 31;
  int f    = idx >> 9;
  int kc   = f % Kc;
  int mt   = f / Kc;
  int row  = mt * 16 + (lane & 15);
  int K    = kc * 32 + ((lane >> 4) * 16) + h;
  out[idx] = (bf16)w[row * Cin + K];
}

// ---------------- pooled features: xp = leaky_relu(mean_8x8(x)) ----------------
__global__ void pool_kernel(const float* __restrict__ x, float* __restrict__ xp) {
  int idx = blockIdx.x * blockDim.x + threadIdx.x;   // B*C*16*16 = 524288
  int wx = idx & 15, wy = (idx >> 4) & 15, c = (idx >> 8) & 255, b = idx >> 16;
  const float* px = x + (((size_t)(b * C_ + c) * H_) + wy * WS_) * W_ + wx * WS_;
  float s = 0.f;
#pragma unroll
  for (int iy = 0; iy < WS_; ++iy)
#pragma unroll
    for (int ix = 0; ix < WS_; ++ix) s += px[iy * W_ + ix];
  s *= (1.f / 64.f);
  xp[idx] = s > 0.f ? s : 0.01f * s;
}

// ---------------- per-window offset/scale prediction ----------------
__global__ void offsets_kernel(const float* __restrict__ xp,
                               const float* __restrict__ bias_w, const float* __restrict__ bias_b,
                               const float* __restrict__ scale_w, const float* __restrict__ scale_b,
                               float* __restrict__ bx, float* __restrict__ by,
                               float* __restrict__ sx, float* __restrict__ sy) {
  int idx = blockIdx.x * blockDim.x + threadIdx.x;   // N_*NWIN_ = 16384
  int win = idx & 255;
  int n   = idx >> 8;
  int head = n & 7, b = n >> 3;
  const float* p = xp + (size_t)b * C_ * NWIN_ + win;  // stride NWIN_ per channel
  const float* bw0 = bias_w + (2 * head) * C_;
  const float* bw1 = bias_w + (2 * head + 1) * C_;
  const float* sw0 = scale_w + (2 * head) * C_;
  const float* sw1 = scale_w + (2 * head + 1) * C_;
  float a0 = 0.f, a1 = 0.f, a2 = 0.f, a3 = 0.f;
  for (int c = 0; c < C_; ++c) {
    float v = p[(size_t)c * NWIN_];
    a0 += bw0[c] * v; a1 += bw1[c] * v;
    a2 += sw0[c] * v; a3 += sw1[c] * v;
  }
  bx[idx] = (a0 + bias_b[2 * head])     * (1.f / (float)WN_);
  by[idx] = (a1 + bias_b[2 * head + 1]) * (1.f / (float)WN_);
  sx[idx] =  a2 + scale_b[2 * head];
  sy[idx] =  a3 + scale_b[2 * head + 1];
}

// ---------------- x transpose: f32 [b][c][p] -> bf16 [b][p][c] ----------------
__global__ void xT_kernel(const float* __restrict__ x, bf16* __restrict__ xT) {
  __shared__ float tile[32][33];
  int b = blockIdx.z, c0 = blockIdx.y * 32, p0 = blockIdx.x * 32;
  for (int i = threadIdx.y; i < 32; i += 8)
    tile[i][threadIdx.x] = x[((size_t)b * C_ + c0 + i) * HW_ + p0 + threadIdx.x];
  __syncthreads();
  for (int i = threadIdx.y; i < 32; i += 8)
    xT[((size_t)b * HW_ + p0 + i) * C_ + c0 + threadIdx.x] = (bf16)tile[threadIdx.x][i];
}

// ---------------- WMMA GEMM: out[b][m][p] = Wswz[m,:] . Bt[b][p][:] + bias[m] ----------------
// One wave per (b, mtile, 64-pixel group). KC is compile time so the 2-deep software
// pipeline fully unrolls: the kc+1 guard folds, buffer rotation becomes SSA renaming,
// and all loads are hoisted ahead of the WMMA stream (staggered s_wait_loadcnt).
template <int KC>
__global__ void __launch_bounds__(256)
wmma_gemm_kernel(const bf16* __restrict__ Aswz, const bf16* __restrict__ Bt,
                 const float* __restrict__ bias, f16* __restrict__ out16,
                 float* __restrict__ out32, int Mtiles) {
  const int Cin = KC * 32;
  int wave = (blockIdx.x * blockDim.x + threadIdx.x) >> 5;
  int lane = threadIdx.x & 31;
  const int npg = HW_ / 64;                 // 256 pixel groups
  int pg = wave % npg;
  int mt = (wave / npg) % Mtiles;
  int b  = wave / (npg * Mtiles);
  int p0 = pg * 64;

  v8f acc[4];
#pragma unroll
  for (int t = 0; t < 4; ++t) acc[t] = zero8();

  const bf16* afr  = Aswz + (size_t)mt * KC * 512;
  const bf16* brow = Bt + (size_t)b * HW_ * Cin;
  const bf16* bbase[4];
#pragma unroll
  for (int t = 0; t < 4; ++t)
    bbase[t] = brow + (size_t)(p0 + t * 16 + (lane & 15)) * Cin + (lane >> 4) * 16;

  // prologue: fragments for kc = 0
  v16bf a_cur = *(const v16bf*)(afr + lane * 16);
  v16bf b_cur[4];
#pragma unroll
  for (int t = 0; t < 4; ++t) b_cur[t] = *(const v16bf*)(bbase[t]);

#pragma unroll
  for (int kc = 0; kc < KC; ++kc) {
    v16bf a_nxt, b_nxt[4];
    if (kc + 1 < KC) {   // compile-time after unroll
      a_nxt = *(const v16bf*)(afr + (kc + 1) * 512 + lane * 16);
#pragma unroll
      for (int t = 0; t < 4; ++t)
        b_nxt[t] = *(const v16bf*)(bbase[t] + (kc + 1) * 32);
    }
#pragma unroll
    for (int t = 0; t < 4; ++t) acc[t] = wmma_bf16(a_cur, b_cur[t], acc[t]);
    a_cur = a_nxt;
#pragma unroll
    for (int t = 0; t < 4; ++t) b_cur[t] = b_nxt[t];
  }

  int g = lane >> 4, nn = lane & 15;
#pragma unroll
  for (int t = 0; t < 4; ++t) {
    int p = p0 + t * 16 + nn;
#pragma unroll
    for (int r = 0; r < 8; ++r) {
      int m = mt * 16 + r + 8 * g;
      float v = acc[t][r] + bias[m];
      if (out16) out16[((size_t)b * (Mtiles * 16) + m) * HW_ + p] = (f16)v;
      else       out32[((size_t)b * (Mtiles * 16) + m) * HW_ + p] = v;
    }
  }
}

// ---------------- grid-sample K/V + window-gather Q into [b][win][head][q][d] f16 ----------------
__device__ inline void corner_w(int yi, int xi, float w, int& off, float& wgt) {
  bool valid = (yi >= 0) && (yi < H_) && (xi >= 0) && (xi < W_);
  int yc = min(max(yi, 0), H_ - 1), xc = min(max(xi, 0), W_ - 1);
  off = yc * W_ + xc;
  wgt = valid ? w : 0.f;
}

__global__ void sample_kernel(const f16* __restrict__ qkv,
                              const float* __restrict__ bx, const float* __restrict__ by,
                              const float* __restrict__ sx, const float* __restrict__ sy,
                              f16* __restrict__ qw, f16* __restrict__ kw, f16* __restrict__ vw) {
  int idx = blockIdx.x * blockDim.x + threadIdx.x;   // N_*NWIN_*WSQ_ = 1048576
  int qpos = idx & 63;
  int win  = (idx >> 6) & 255;
  int n    = idx >> 14;
  int head = n & 7, b = n >> 3;
  int wx = win & 15, wy = win >> 4;
  int ix = qpos & 7, iy = qpos >> 3;
  int oidx = (n << 8) | win;

  const float inv = 2.0f / 127.0f;
  float offw = ((float)ix - 3.5f) * inv;
  float offh = ((float)iy - 3.5f) * inv;
  float cx = (-1.f + ((float)(wx * 8) + 3.5f) * inv) + offw + offw * sx[oidx] + bx[oidx];
  float cy = (-1.f + ((float)(wy * 8) + 3.5f) * inv) + offh + offh * sy[oidx] + by[oidx];
  float gx = (cx + 1.f) * 0.5f * 127.f;
  float gy = (cy + 1.f) * 0.5f * 127.f;
  float fx0 = floorf(gx), fy0 = floorf(gy);
  int x0 = (int)fx0, y0 = (int)fy0;
  float wx1 = gx - fx0, wx0m = 1.f - wx1;
  float wy1 = gy - fy0, wy0m = 1.f - wy1;

  int o00, o01, o10, o11; float w00, w01, w10, w11;
  corner_w(y0,     x0,     wy0m * wx0m, o00, w00);
  corner_w(y0,     x0 + 1, wy0m * wx1,  o01, w01);
  corner_w(y0 + 1, x0,     wy1  * wx0m, o10, w10);
  corner_w(y0 + 1, x0 + 1, wy1  * wx1,  o11, w11);

  const f16* qbase = qkv + ((size_t)b * 768 +       head * HD_) * HW_;
  const f16* kbase = qkv + ((size_t)b * 768 + 256 + head * HD_) * HW_;
  const f16* vbase = qkv + ((size_t)b * 768 + 512 + head * HD_) * HW_;
  int p = (wy * 8 + iy) * W_ + wx * 8 + ix;
  size_t wbase = ((((size_t)b * NWIN_ + win) * NH_ + head) * WSQ_ + qpos) * HD_;

  for (int d = 0; d < HD_; ++d) {
    const f16* kd = kbase + (size_t)d * HW_;
    const f16* vd = vbase + (size_t)d * HW_;
    float kv = w00 * (float)kd[o00] + w01 * (float)kd[o01] +
               w10 * (float)kd[o10] + w11 * (float)kd[o11];
    float vv = w00 * (float)vd[o00] + w01 * (float)vd[o01] +
               w10 * (float)vd[o10] + w11 * (float)vd[o11];
    kw[wbase + d] = (f16)kv;
    vw[wbase + d] = (f16)vv;
    qw[wbase + d] = qbase[(size_t)d * HW_ + p];
  }
}

// ---------------- window attention: one wave per (b, win, head) ----------------
// Score stage is tiled by 16-row M-block so only 4 S-fragments (32 VGPRs) are live
// at a time: keeps allocation under 256 VGPRs (no VGPR-MSB mode / copy storms) and
// interleaves WMMA with softmax VALU per block.
__global__ void __launch_bounds__(128)
attn_kernel(const f16* __restrict__ qw, const f16* __restrict__ kw,
            const f16* __restrict__ vw, const float* __restrict__ rpb_table,
            bf16* __restrict__ yT) {
  __shared__ f16 lds[4 * 6144];   // per wave: P[64][64] + VT[32][64]
  int wid  = threadIdx.x >> 5;
  int lane = threadIdx.x & 31;
  int gw   = blockIdx.x * 4 + wid;           // B_*NWIN_*NH_ = 16384
  int head = gw & 7;
  int win  = (gw >> 3) & 255;
  int b    = gw >> 11;

  f16* P  = lds + wid * 6144;
  f16* VT = P + 4096;

  size_t base = (((size_t)b * NWIN_ + win) * NH_ + head) * WSQ_ * HD_;
  const f16* Q = qw + base;
  const f16* K = kw + base;
  const f16* V = vw + base;

  // stage V transposed: VT[d][q]
  for (int i = lane; i < WSQ_ * HD_; i += 32) {
    int d = i & 31, q = i >> 5;
    VT[d * WSQ_ + q] = V[q * HD_ + d];
  }

  const float scale = 0.17677669529663687f;  // 32^-0.5
  int g = lane >> 4, nn = lane & 15;

  // K^T B-fragments (shared by all M-blocks)
  v16h bfr[4];
#pragma unroll
  for (int nt = 0; nt < 4; ++nt) bfr[nt] = loadBT16(K + nt * 16 * HD_, HD_, lane);

  // Per 16-row M-block: 4 WMMAs -> softmax -> P rows to LDS
#pragma unroll
  for (int mt = 0; mt < 4; ++mt) {
    v16h a = loadA16(Q + mt * 16 * HD_, HD_, lane);
    v8f S[4];
#pragma unroll
    for (int nt = 0; nt < 4; ++nt) S[nt] = wmma_f16(a, bfr[nt], zero8());

#pragma unroll
    for (int r = 0; r < 8; ++r) {
      int m = mt * 16 + r + 8 * g;
      int qy = m >> 3, qx = m & 7;
      float vals[4];
      float mx = -1e30f;
#pragma unroll
      for (int nt = 0; nt < 4; ++nt) {
        int kp = nt * 16 + nn;
        int ky = kp >> 3, kx = kp & 7;
        int ridx = (qy - ky + 7) * 15 + (qx - kx + 7);
        float v = S[nt][r] * scale + rpb_table[ridx * NH_ + head];
        vals[nt] = v;
        mx = fmaxf(mx, v);
      }
#pragma unroll
      for (int o = 1; o < 16; o <<= 1) mx = fmaxf(mx, __shfl_xor(mx, o, 32));
      float sum = 0.f;
#pragma unroll
      for (int nt = 0; nt < 4; ++nt) { vals[nt] = __expf(vals[nt] - mx); sum += vals[nt]; }
#pragma unroll
      for (int o = 1; o < 16; o <<= 1) sum += __shfl_xor(sum, o, 32);
      float inv = 1.f / sum;
#pragma unroll
      for (int nt = 0; nt < 4; ++nt)
        P[m * WSQ_ + nt * 16 + nn] = (f16)(vals[nt] * inv);
    }
  }
  asm volatile("s_wait_dscnt 0" ::: "memory");

  // O = P . V  (16 WMMAs, K = 64 in two chunks)
  v8f O[4][2];
#pragma unroll
  for (int mt = 0; mt < 4; ++mt)
#pragma unroll
    for (int nt = 0; nt < 2; ++nt) O[mt][nt] = zero8();

#pragma unroll
  for (int kc = 0; kc < 2; ++kc) {
    v16h pa[4], vb[2];
#pragma unroll
    for (int mt = 0; mt < 4; ++mt) pa[mt] = loadA16(P + mt * 16 * WSQ_ + kc * 32, WSQ_, lane);
#pragma unroll
    for (int nt = 0; nt < 2; ++nt) vb[nt] = loadBT16(VT + nt * 16 * WSQ_ + kc * 32, WSQ_, lane);
#pragma unroll
    for (int mt = 0; mt < 4; ++mt)
#pragma unroll
      for (int nt = 0; nt < 2; ++nt) O[mt][nt] = wmma_f16(pa[mt], vb[nt], O[mt][nt]);
  }

  // write pre-projection activations pixel-major: yT[b][p][head*32+d] bf16
  int wy = win >> 4, wx = win & 15;
#pragma unroll
  for (int mt = 0; mt < 4; ++mt) {
#pragma unroll
    for (int r = 0; r < 8; ++r) {
      int m = mt * 16 + r + 8 * g;
      int p = (wy * 8 + (m >> 3)) * W_ + wx * 8 + (m & 7);
#pragma unroll
      for (int nt = 0; nt < 2; ++nt) {
        int c = head * HD_ + nt * 16 + nn;
        yT[((size_t)b * HW_ + p) * C_ + c] = (bf16)O[mt][nt][r];
      }
    }
  }
}

// ---------------- host launcher ----------------
extern "C" void kernel_launch(void* const* d_in, const int* in_sizes, int n_in,
                              void* d_out, int out_size, void* d_ws, size_t ws_size,
                              hipStream_t stream) {
  const float* x        = (const float*)d_in[0];
  const float* qkv_w    = (const float*)d_in[1];
  const float* qkv_b    = (const float*)d_in[2];
  const float* bias_w   = (const float*)d_in[3];
  const float* bias_b   = (const float*)d_in[4];
  const float* scale_w  = (const float*)d_in[5];
  const float* scale_b  = (const float*)d_in[6];
  const float* proj_w   = (const float*)d_in[7];
  const float* proj_b   = (const float*)d_in[8];
  const float* rpb      = (const float*)d_in[9];
  float* out = (float*)d_out;

  char* ws = (char*)d_ws;
  size_t off = 0;
  auto alloc = [&](size_t bytes) -> void* {
    void* p = ws + off;
    off = (off + bytes + 255) & ~(size_t)255;
    return p;
  };

  bf16* wAq = (bf16*)alloc((size_t)48 * 8 * 512 * 2);
  bf16* wAp = (bf16*)alloc((size_t)16 * 8 * 512 * 2);
  float* xp = (float*)alloc((size_t)B_ * C_ * NWIN_ * 4);
  float* bx = (float*)alloc((size_t)N_ * NWIN_ * 4);
  float* by = (float*)alloc((size_t)N_ * NWIN_ * 4);
  float* sx = (float*)alloc((size_t)N_ * NWIN_ * 4);
  float* sy = (float*)alloc((size_t)N_ * NWIN_ * 4);
  bf16* xT  = (bf16*)alloc((size_t)B_ * HW_ * C_ * 2);
  f16* qkv  = (f16*) alloc((size_t)B_ * 768 * HW_ * 2);
  f16* qw   = (f16*) alloc((size_t)B_ * NWIN_ * NH_ * WSQ_ * HD_ * 2);
  f16* kw   = (f16*) alloc((size_t)B_ * NWIN_ * NH_ * WSQ_ * HD_ * 2);
  f16* vw   = (f16*) alloc((size_t)B_ * NWIN_ * NH_ * WSQ_ * HD_ * 2);
  bf16* yT  = (bf16*)alloc((size_t)B_ * HW_ * C_ * 2);

  swizzle_w_kernel<<<(48 * 8 * 512 + 255) / 256, 256, 0, stream>>>(qkv_w, wAq, 48, 8, C_);
  swizzle_w_kernel<<<(16 * 8 * 512 + 255) / 256, 256, 0, stream>>>(proj_w, wAp, 16, 8, C_);
  pool_kernel<<<(B_ * C_ * NWIN_) / 256, 256, 0, stream>>>(x, xp);
  offsets_kernel<<<(N_ * NWIN_) / 256, 256, 0, stream>>>(xp, bias_w, bias_b, scale_w, scale_b,
                                                         bx, by, sx, sy);
  dim3 tg(HW_ / 32, C_ / 32, B_), tb(32, 8);
  xT_kernel<<<tg, tb, 0, stream>>>(x, xT);

  // QKV GEMM: 768x256 over all pixels, bf16 WMMA (KC=8 compile-time)
  wmma_gemm_kernel<8><<<(B_ * 48 * 256) / 8, 256, 0, stream>>>(wAq, xT, qkv_b, qkv, nullptr, 48);

  sample_kernel<<<(N_ * NWIN_ * WSQ_) / 256, 256, 0, stream>>>(qkv, bx, by, sx, sy, qw, kw, vw);

  attn_kernel<<<(B_ * NWIN_ * NH_) / 4, 128, 0, stream>>>(qw, kw, vw, rpb, yT);

  // output projection: 256x256, f32 out (KC=8 compile-time)
  wmma_gemm_kernel<8><<<(B_ * 16 * 256) / 8, 256, 0, stream>>>(wAp, yT, proj_b, nullptr, out, 16);
}
